// HeteroGAT_64570538328497
// MI455X (gfx1250) — matrix operand order repaired and checked by reading, not compiled, not run on hardware
//
#include <hip/hip_runtime.h>
#include <hip/hip_bf16.h>

// ---------------------------------------------------------------------------
// 2-layer GAT for gfx1250 (MI455X).  GEMMs via v_wmma_f32_16x16x32_f16.
// ---------------------------------------------------------------------------

typedef __attribute__((ext_vector_type(16))) _Float16 v16h;
typedef __attribute__((ext_vector_type(8)))  _Float16 v8h;
typedef __attribute__((ext_vector_type(8)))  float    v8f;

#define NEG_SLOPE 0.2f

__device__ __forceinline__ float lrelu(float x) { return x > 0.f ? x : NEG_SLOPE * x; }

// Monotonic float atomic-max via sign-split int/uint atomics. Init to -inf.
__device__ __forceinline__ void atomicMaxFloat(float* addr, float value) {
    if (value >= 0.f) atomicMax((int*)addr, __float_as_int(value));
    else              atomicMin((unsigned int*)addr, __float_as_uint(value));
}

// ---------------- elementwise conversions -----------------------------------

__global__ void cvt_f32_to_f16_kernel(const float* __restrict__ src,
                                      _Float16* __restrict__ dst, int n) {
    int i = blockIdx.x * blockDim.x + threadIdx.x;
    if (i < n) dst[i] = (_Float16)src[i];
}

// W row-major [K, Ncol] fp32 -> column-major fp16 Wcm[n*K + k]
__global__ void cvt_w_colmajor_kernel(const float* __restrict__ W,
                                      _Float16* __restrict__ Wcm,
                                      int K, int Ncol) {
    int i = blockIdx.x * blockDim.x + threadIdx.x;
    if (i >= K * Ncol) return;
    int k = i / Ncol, n = i - k * Ncol;
    Wcm[(size_t)n * K + k] = (_Float16)W[i];
}

// ---------------- WMMA GEMM: C[M,Ncol] = A[M,K] * W[K,Ncol] ------------------
// A fp16 row-major, B fp16 column-major (Bcm[n*K+k] = W[k,n]), C fp32.
// One wave per 16x16 output tile.  M % 16 == 0, K % 32 == 0, Ncol % 16 == 0.
__global__ void gemm_wmma_f16_kernel(const _Float16* __restrict__ A,
                                     const _Float16* __restrict__ Bcm,
                                     float* __restrict__ C,
                                     int M, int K, int Ncol,
                                     int tiles_n, int total_tiles) {
    int wid = (int)((blockIdx.x * blockDim.x + threadIdx.x) >> 5);
    if (wid >= total_tiles) return;            // uniform per wave -> EXEC all-1s below
    int lane = threadIdx.x & 31;
    int tm = wid / tiles_n, tn = wid - tm * tiles_n;
    int r15 = lane & 15;                        // row (A) / col (B) / col (C)
    int hi  = lane >> 4;                        // lane-half selects K sub-block

    const _Float16* arow = A   + (size_t)(tm * 16 + r15) * K;
    const _Float16* bcol = Bcm + (size_t)(tn * 16 + r15) * K;

    v8f acc = {};
    for (int k0 = 0; k0 < K; k0 += 32) {
        // A 16x32 fragment: elems 0..7 = K k0+hi*8..+7, elems 8..15 = K k0+16+hi*8..+7
        v8h alo = *(const v8h*)(arow + k0 + hi * 8);
        v8h ahi = *(const v8h*)(arow + k0 + 16 + hi * 8);
        v16h a;
#pragma unroll
        for (int i = 0; i < 8; ++i) { a[i] = alo[i]; a[8 + i] = ahi[i]; }
        // B 32x16 fragment: elems 0..7 = K k0+hi*8..+7, elems 8..15 = K k0+16+hi*8..+7
        v8h blo = *(const v8h*)(bcol + k0 + hi * 8);
        v8h bhi = *(const v8h*)(bcol + k0 + 16 + hi * 8);
        v16h b;
#pragma unroll
        for (int i = 0; i < 8; ++i) { b[i] = blo[i]; b[8 + i] = bhi[i]; }
        acc = __builtin_amdgcn_wmma_f32_16x16x32_f16(false, a, false, b,
                                                     (short)0, acc, false, false);
    }
    // C/D layout: VGPR r -> row r + 8*hi ; col = lane&15
    int col = tn * 16 + r15;
#pragma unroll
    for (int r = 0; r < 8; ++r) {
        int row = tm * 16 + r + 8 * hi;
        C[(size_t)row * Ncol + col] = acc[r];
    }
}

// ---------------- per-node attention scores ---------------------------------
// es[n,h] = dot(h[n,h,:], a_src[h,:]); ed likewise.  Also init emax/denom.
__global__ void node_scores_kernel(const float* __restrict__ h,
                                   const float* __restrict__ a_src,
                                   const float* __restrict__ a_dst,
                                   float* __restrict__ es, float* __restrict__ ed,
                                   float* __restrict__ emax, float* __restrict__ denom,
                                   int N, int H, int C) {
    int tid = blockIdx.x * blockDim.x + threadIdx.x;
    if (tid >= N * H) return;
    int n = tid / H, hd = tid - n * H;
    const float* hp = h + ((size_t)n * H + hd) * C;
    float s = 0.f, d = 0.f;
    for (int c = 0; c < C; ++c) {
        float v = hp[c];
        s += v * a_src[hd * C + c];
        d += v * a_dst[hd * C + c];
    }
    es[tid] = s; ed[tid] = d;
    emax[tid]  = -__builtin_inff();
    denom[tid] = 0.f;
}

// ---------------- edge passes ------------------------------------------------
__device__ __forceinline__ void edge_endpoints(const long long* __restrict__ ei,
                                               int e, int E, int& s, int& d) {
    if (e < E) { s = (int)ei[e]; d = (int)ei[(size_t)E + e]; }
    else       { s = d = e - E; }                     // self loops appended
}

__global__ void edge_max_kernel(const long long* __restrict__ ei, int E, int E2,
                                const float* __restrict__ es, const float* __restrict__ ed,
                                float* __restrict__ emax, int H) {
    int tid = blockIdx.x * blockDim.x + threadIdx.x;
    if (tid >= E2 * H) return;
    int e = tid / H, hd = tid - e * H;
    int s, d; edge_endpoints(ei, e, E, s, d);
    float v = lrelu(es[(size_t)s * H + hd] + ed[(size_t)d * H + hd]);
    atomicMaxFloat(&emax[(size_t)d * H + hd], v);
}

__global__ void edge_denom_kernel(const long long* __restrict__ ei, int E, int E2,
                                  const float* __restrict__ es, const float* __restrict__ ed,
                                  const float* __restrict__ emax, float* __restrict__ denom,
                                  int H) {
    int tid = blockIdx.x * blockDim.x + threadIdx.x;
    if (tid >= E2 * H) return;
    int e = tid / H, hd = tid - e * H;
    int s, d; edge_endpoints(ei, e, E, s, d);
    float v = lrelu(es[(size_t)s * H + hd] + ed[(size_t)d * H + hd]);
    float p = __expf(v - emax[(size_t)d * H + hd]);
    atomicAdd(&denom[(size_t)d * H + hd], p);
}

// One wave32 per (edge, head); lane == channel (C == 32 == wave width).
__global__ void edge_aggregate_kernel(const long long* __restrict__ ei, int E, int E2,
                                      const float* __restrict__ es, const float* __restrict__ ed,
                                      const float* __restrict__ emax, const float* __restrict__ denom,
                                      const float* __restrict__ h, float* __restrict__ agg,
                                      int H) {
    long long t = (long long)blockIdx.x * blockDim.x + threadIdx.x;
    int wid = (int)(t >> 5);
    if (wid >= E2 * H) return;
    int lane = threadIdx.x & 31;                      // channel
    int e = wid / H, hd = wid - e * H;
    int s, d; edge_endpoints(ei, e, E, s, d);
    float v = lrelu(es[(size_t)s * H + hd] + ed[(size_t)d * H + hd]);
    float alpha = __expf(v - emax[(size_t)d * H + hd]) /
                  (denom[(size_t)d * H + hd] + 1e-16f);
    float val = h[((size_t)s * H + hd) * 32 + lane] * alpha;
    atomicAdd(&agg[((size_t)d * H + hd) * 32 + lane], val);
}

// ---------------- epilogues --------------------------------------------------
// Layer-1 epilogue: out = relu(agg + bias) -> fp16 input for layer-2 GEMM.
__global__ void bias_relu_f16_kernel(const float* __restrict__ agg,
                                     const float* __restrict__ bias,
                                     _Float16* __restrict__ out_h,
                                     int n, int fmask) {
    int i = blockIdx.x * blockDim.x + threadIdx.x;
    if (i >= n) return;
    float v = agg[i] + bias[i & fmask];
    v = v > 0.f ? v : 0.f;
    out_h[i] = (_Float16)v;
}

// Final: d_out = agg2 + b2
__global__ void final_bias_kernel(const float* __restrict__ agg,
                                  const float* __restrict__ bias,
                                  float* __restrict__ out, int n, int fmask) {
    int i = blockIdx.x * blockDim.x + threadIdx.x;
    if (i >= n) return;
    out[i] = agg[i] + bias[i & fmask];
}

// ---------------------------------------------------------------------------

static inline int ceil_div(long long a, long long b) { return (int)((a + b - 1) / b); }

extern "C" void kernel_launch(void* const* d_in, const int* in_sizes, int n_in,
                              void* d_out, int out_size, void* d_ws, size_t ws_size,
                              hipStream_t stream) {
    const float*      x      = (const float*)d_in[0];
    const long long*  ei     = (const long long*)d_in[1];
    const float*      W1     = (const float*)d_in[2];
    const float*      a_src1 = (const float*)d_in[3];
    const float*      a_dst1 = (const float*)d_in[4];
    const float*      b1     = (const float*)d_in[5];
    const float*      W2     = (const float*)d_in[6];
    const float*      a_src2 = (const float*)d_in[7];
    const float*      a_dst2 = (const float*)d_in[8];
    const float*      b2     = (const float*)d_in[9];

    const int IN = 128, H1 = 2, C = 32, HC1 = H1 * C;   // layer1: 128 -> 2x32
    const int K2 = HC1, C2 = 32;                        // layer2: 64 -> 1x32
    const int N  = in_sizes[0] / IN;
    const int E  = in_sizes[1] / 2;
    const int E2 = E + N;                               // with self loops

    // ---- carve scratch ----
    char* p = (char*)d_ws;
    auto alloc = [&](size_t bytes) { char* r = p; p += (bytes + 255) & ~(size_t)255; return r; };
    _Float16* xh    = (_Float16*)alloc((size_t)N * IN * sizeof(_Float16));
    _Float16* w1cm  = (_Float16*)alloc((size_t)IN * HC1 * sizeof(_Float16));
    float*    h1    = (float*)alloc((size_t)N * HC1 * sizeof(float));
    float*    es1   = (float*)alloc((size_t)N * H1 * sizeof(float));
    float*    ed1   = (float*)alloc((size_t)N * H1 * sizeof(float));
    float*    emax1 = (float*)alloc((size_t)N * H1 * sizeof(float));
    float*    den1  = (float*)alloc((size_t)N * H1 * sizeof(float));
    float*    agg1  = (float*)alloc((size_t)N * HC1 * sizeof(float));
    _Float16* h1h   = (_Float16*)alloc((size_t)N * HC1 * sizeof(_Float16));
    _Float16* w2cm  = (_Float16*)alloc((size_t)K2 * C2 * sizeof(_Float16));
    float*    h2    = (float*)alloc((size_t)N * C2 * sizeof(float));
    float*    es2   = (float*)alloc((size_t)N * sizeof(float));
    float*    ed2   = (float*)alloc((size_t)N * sizeof(float));
    float*    emax2 = (float*)alloc((size_t)N * sizeof(float));
    float*    den2  = (float*)alloc((size_t)N * sizeof(float));
    float*    agg2  = (float*)alloc((size_t)N * C2 * sizeof(float));
    (void)ws_size; (void)n_in; (void)out_size;

    const int B = 256;

    // ================= Layer 1 =================
    cvt_f32_to_f16_kernel<<<ceil_div((long long)N * IN, B), B, 0, stream>>>(x, xh, N * IN);
    cvt_w_colmajor_kernel<<<ceil_div((long long)IN * HC1, B), B, 0, stream>>>(W1, w1cm, IN, HC1);

    {
        int tiles_n = HC1 / 16, total = (N / 16) * tiles_n;
        gemm_wmma_f16_kernel<<<ceil_div((long long)total * 32, B), B, 0, stream>>>(
            xh, w1cm, h1, N, IN, HC1, tiles_n, total);
    }

    node_scores_kernel<<<ceil_div((long long)N * H1, B), B, 0, stream>>>(
        h1, a_src1, a_dst1, es1, ed1, emax1, den1, N, H1, C);
    hipMemsetAsync(agg1, 0, (size_t)N * HC1 * sizeof(float), stream);

    edge_max_kernel<<<ceil_div((long long)E2 * H1, B), B, 0, stream>>>(ei, E, E2, es1, ed1, emax1, H1);
    edge_denom_kernel<<<ceil_div((long long)E2 * H1, B), B, 0, stream>>>(ei, E, E2, es1, ed1, emax1, den1, H1);
    edge_aggregate_kernel<<<ceil_div((long long)E2 * H1 * 32, B), B, 0, stream>>>(
        ei, E, E2, es1, ed1, emax1, den1, h1, agg1, H1);

    bias_relu_f16_kernel<<<ceil_div((long long)N * HC1, B), B, 0, stream>>>(
        agg1, b1, h1h, N * HC1, HC1 - 1);

    // ================= Layer 2 =================
    cvt_w_colmajor_kernel<<<ceil_div((long long)K2 * C2, B), B, 0, stream>>>(W2, w2cm, K2, C2);
    {
        int tiles_n = C2 / 16, total = (N / 16) * tiles_n;
        gemm_wmma_f16_kernel<<<ceil_div((long long)total * 32, B), B, 0, stream>>>(
            h1h, w2cm, h2, N, K2, C2, tiles_n, total);
    }

    node_scores_kernel<<<ceil_div((long long)N, B), B, 0, stream>>>(
        h2, a_src2, a_dst2, es2, ed2, emax2, den2, N, 1, C2);
    hipMemsetAsync(agg2, 0, (size_t)N * C2 * sizeof(float), stream);

    edge_max_kernel<<<ceil_div((long long)E2, B), B, 0, stream>>>(ei, E, E2, es2, ed2, emax2, 1);
    edge_denom_kernel<<<ceil_div((long long)E2, B), B, 0, stream>>>(ei, E, E2, es2, ed2, emax2, den2, 1);
    edge_aggregate_kernel<<<ceil_div((long long)E2 * 32, B), B, 0, stream>>>(
        ei, E, E2, es2, ed2, emax2, den2, h2, agg2, 1);

    final_bias_kernel<<<ceil_div((long long)N * C2, B), B, 0, stream>>>(
        agg2, b2, (float*)d_out, N * C2, C2 - 1);
}
